// GraphEncoder_56418690400481
// MI455X (gfx1250) — matrix operand order repaired and checked by reading, not compiled, not run on hardware
//
#include <hip/hip_runtime.h>
#include <hip/hip_bf16.h>

// ---------------------------------------------------------------------------
// GraphEncoder: fused GRU (WMMA f16 16x16x32, TDM-pipelined staging)
//               + GAT segment-softmax
// Target: gfx1250 (CDNA5, wave32, WMMA, Tensor Data Mover)
// ---------------------------------------------------------------------------

typedef __attribute__((ext_vector_type(16))) _Float16     v16h;
typedef __attribute__((ext_vector_type(8)))  float        v8f;
typedef __attribute__((ext_vector_type(4)))  unsigned int u32x4;
typedef __attribute__((ext_vector_type(8)))  int          i32x8;
typedef __attribute__((ext_vector_type(4)))  int          i32x4;

#define T_STEPS 16
#define D_IN    128
#define H_DIM   128
#define G3      (3 * H_DIM)          // 384
#define MT_PER_BLK 4                 // 4 M-tiles of 16 rows -> 64 nodes / block
#define NODES_PER_BLK (MT_PER_BLK * 16)
#define FRAG_ELEMS 512               // 32 lanes * 16 halves per 16x32 fragment
#define NEG_SLOPE 0.2f

#if defined(__HIP_DEVICE_COMPILE__) && __has_builtin(__builtin_amdgcn_tensor_load_to_lds)
#define USE_TDM 1
#else
#define USE_TDM 0
#endif

// ---- WMMA helper ----------------------------------------------------------
__device__ __forceinline__ v8f wmma_f16(v16h a, v16h b, v8f c) {
    // (neg_a, A, neg_b, B, c_mod, C, reuse_a, reuse_b)
    return __builtin_amdgcn_wmma_f32_16x16x32_f16(false, a, false, b,
                                                  (short)0, c, false, false);
}

// ---- fragment index helpers (CDNA5 ISA 7.12.2, wave32) --------------------
// A matrix 16x32 f16: lane = m + 16*((k%16)>=8), half-pos = (k&7) | 8*(k>=16)
__device__ __forceinline__ int afrag_idx(int m, int kk) {
    int lane = m + ((kk & 8) ? 16 : 0);
    int pos  = (kk & 7) | ((kk & 16) ? 8 : 0);
    return lane * 16 + pos;
}
// B matrix 32x16 f16 (stores W^T): lane = n + 16*((k%16)>=8), pos = (k&7) | 8*(k>=16)
__device__ __forceinline__ int bfrag_idx(int u, int d) {
    int ut = u >> 4, n = u & 15;
    int kf = d >> 5, kk = d & 31;
    int lane = n + ((kk & 8) ? 16 : 0);
    int pos  = (kk & 7) | ((kk & 16) ? 8 : 0);
    return ((ut * 4 + kf) * 32 + lane) * 16 + pos;
}

// ---- TDM: async 2-D tile DMA (global -> LDS), tracked by TENSORcnt --------
#if USE_TDM
__device__ __forceinline__ void tdm_load_tile(unsigned lds_off,
                                              unsigned long long gaddr,
                                              int valid_rows) {
    // D# group 0: count=1 (valid), lds_addr, global_addr, type=2 (image)
    u32x4 g0 = {
        1u,                                             // count=1, user mode
        lds_off,                                        // LDS byte address
        (unsigned)(gaddr & 0xFFFFFFFFull),              // global_addr[31:0]
        (unsigned)((gaddr >> 32) & 0x1FFFFFFu) | (2u << 30) // [56:32] | type=2
    };
    // D# group 1: data_size=4B, tensor 128 x valid_rows (stride T*D),
    //             tile 128 x 64 -> OOB rows are zero-filled by TDM.
    i32x8 g1 = {
        (int)(2u << 16),                 // data_size = 4 bytes
        (int)(128u << 16),               // tensor_dim0 = 128 (bits 63:48)
        (int)((unsigned)valid_rows << 16), // tensor_dim1[15:0] (bits 95:80)
        (int)(128u << 16),               // tile_dim0 = 128 (bits 127:112)
        64,                              // tile_dim1 = 64 rows
        T_STEPS * D_IN,                  // tensor_dim0_stride = 2048 elems
        0, 0
    };
    i32x4 z4 = {0, 0, 0, 0};
#if __clang_major__ >= 23
    i32x8 z8 = {0, 0, 0, 0, 0, 0, 0, 0};
    __builtin_amdgcn_tensor_load_to_lds(g0, g1, z4, z4, z8, 0);
#else
    __builtin_amdgcn_tensor_load_to_lds(g0, g1, z4, z4, 0);
#endif
}
#endif

// ---- weight pre-swizzle: fp32 row-major W[u][d] -> f16 B-fragments --------
__global__ void k_prep(const float* __restrict__ Wih,
                       const float* __restrict__ Whh,
                       const float* __restrict__ lin,
                       _Float16* __restrict__ fWih,
                       _Float16* __restrict__ fWhh,
                       _Float16* __restrict__ flin) {
    const int NW = G3 * D_IN;        // 49152
    const int NL = H_DIM * H_DIM;    // 16384
    int t = blockIdx.x * 256 + threadIdx.x;
    if (t < NW) {
        int u = t / D_IN, d = t % D_IN;
        fWih[bfrag_idx(u, d)] = (_Float16)Wih[t];
    } else if (t < 2 * NW) {
        int e = t - NW;
        int u = e / D_IN, d = e % D_IN;
        fWhh[bfrag_idx(u, d)] = (_Float16)Whh[e];
    } else if (t < 2 * NW + NL) {
        int e = t - 2 * NW;
        int u = e / H_DIM, d = e % H_DIM;
        flin[bfrag_idx(u, d)] = (_Float16)lin[e];
    }
}

__device__ __forceinline__ float sigmoidf_(float x) {
    return 1.f / (1.f + __expf(-x));
}

// ---- fused GRU + GAT linear ----------------------------------------------
__global__ __launch_bounds__(256)
void k_gru(const float* __restrict__ seq,
           const _Float16* __restrict__ fWih,
           const _Float16* __restrict__ fWhh,
           const _Float16* __restrict__ flin,
           const float* __restrict__ b_ih,
           const float* __restrict__ b_hh,
           float* __restrict__ h_out,      // seq_vector [N,H]
           float* __restrict__ x_out,      // GAT-projected x [N,H]
           int N) {
    __shared__ __align__(32) _Float16 xfrag[MT_PER_BLK * 4 * FRAG_ELEMS]; // 16 KB
    __shared__ __align__(32) _Float16 hfrag[MT_PER_BLK * 4 * FRAG_ELEMS]; // 16 KB
#if USE_TDM
    __shared__ __align__(16) float rawx[2][NODES_PER_BLK * D_IN];         // 64 KB
#endif

    const int tid   = threadIdx.x;
    const int wave  = tid >> 5;      // 8 waves, wave w owns H-cols [16w,16w+16)
    const int lane  = tid & 31;
    const int nhalf = lane >> 4;     // C-layout: M = g + 8*nhalf
    const int ncol  = lane & 15;     // C-layout: N = ncol
    const int ucol  = wave * 16 + ncol;

    // per-lane gate biases (unit column is fixed for this lane)
    const float bir = b_ih[ucol],             bhr = b_hh[ucol];
    const float biz = b_ih[H_DIM + ucol],     bhz = b_hh[H_DIM + ucol];
    const float bin = b_ih[2 * H_DIM + ucol], bhn = b_hh[2 * H_DIM + ucol];

    const long long node_base = (long long)blockIdx.x * NODES_PER_BLK;
    const int rows = (int)min((long long)NODES_PER_BLK, (long long)N - node_base);

#if USE_TDM
    // Kick off the TDM DMA of x_0 while we zero h0.
    const unsigned long long seq_base =
        (unsigned long long)(uintptr_t)seq + (unsigned long long)node_base * T_STEPS * D_IN * 4ull;
    const unsigned raw_off[2] = {
        (unsigned)(uintptr_t)&rawx[0][0],
        (unsigned)(uintptr_t)&rawx[1][0]
    };
    if (wave == 0)
        tdm_load_tile(raw_off[0], seq_base, rows);
#endif

    // h0 = 0
    for (int i = tid; i < MT_PER_BLK * 4 * FRAG_ELEMS; i += 256)
        hfrag[i] = (_Float16)0.f;

    const v8f z8 = {0.f, 0.f, 0.f, 0.f, 0.f, 0.f, 0.f, 0.f};
    v8f h_c[MT_PER_BLK];
#pragma unroll
    for (int mt = 0; mt < MT_PER_BLK; ++mt) h_c[mt] = z8;

    const int hkf = ucol >> 5, hkk = ucol & 31;   // where this lane's h-col lives

    for (int t = 0; t < T_STEPS; ++t) {
#if USE_TDM
        // ---- pipeline: issue DMA for t+1, then wait for tile t ----
        if (wave == 0) {
            if (t + 1 < T_STEPS) {
                tdm_load_tile(raw_off[(t + 1) & 1],
                              seq_base + (unsigned long long)(t + 1) * D_IN * 4ull,
                              rows);
                __builtin_amdgcn_s_wait_tensorcnt(1);   // tile t landed
            } else {
                __builtin_amdgcn_s_wait_tensorcnt(0);
            }
        }
        __syncthreads();   // tile t visible to all waves (+ hfrag from t-1)

        // ---- convert raw fp32 tile -> f16 A-fragments ----
#pragma unroll
        for (int i = 0; i < 8; ++i) {
            int e4  = tid + 256 * i;          // 2048 float4 = 64 rows x 128 d
            int row = e4 >> 5;
            int d0  = (e4 & 31) * 4;
            float4 v = *(const float4*)&rawx[t & 1][row * D_IN + d0];
            int base = ((row >> 4) * 4 + (d0 >> 5)) * FRAG_ELEMS;
            int m = row & 15, kk = d0 & 31;
            xfrag[base + afrag_idx(m, kk + 0)] = (_Float16)v.x;
            xfrag[base + afrag_idx(m, kk + 1)] = (_Float16)v.y;
            xfrag[base + afrag_idx(m, kk + 2)] = (_Float16)v.z;
            xfrag[base + afrag_idx(m, kk + 3)] = (_Float16)v.w;
        }
#else
        // ---- fallback: direct global stage of x_t ----
#pragma unroll
        for (int i = 0; i < 8; ++i) {
            int e4  = tid + 256 * i;
            int row = e4 >> 5;
            int d0  = (e4 & 31) * 4;
            long long node = node_base + row;
            float4 v = make_float4(0.f, 0.f, 0.f, 0.f);
            if (node < N)
                v = *(const float4*)(seq + ((size_t)node * T_STEPS + t) * D_IN + d0);
            int base = ((row >> 4) * 4 + (d0 >> 5)) * FRAG_ELEMS;
            int m = row & 15, kk = d0 & 31;
            xfrag[base + afrag_idx(m, kk + 0)] = (_Float16)v.x;
            xfrag[base + afrag_idx(m, kk + 1)] = (_Float16)v.y;
            xfrag[base + afrag_idx(m, kk + 2)] = (_Float16)v.z;
            xfrag[base + afrag_idx(m, kk + 3)] = (_Float16)v.w;
        }
#endif
        __syncthreads();   // xfrag + hfrag ready

        // ---- 6 GEMM tiles per M-tile, gates in C layout ----
#pragma unroll
        for (int mt = 0; mt < MT_PER_BLK; ++mt) {
            v8f ir = z8, iz = z8, inn = z8, hr = z8, hz = z8, hn = z8;
#pragma unroll
            for (int kf = 0; kf < 4; ++kf) {
                const v16h ax = *(const v16h*)(xfrag + (mt * 4 + kf) * FRAG_ELEMS + lane * 16);
                const v16h ah = *(const v16h*)(hfrag + (mt * 4 + kf) * FRAG_ELEMS + lane * 16);
                v16h b;
                b = *(const v16h*)(fWih + ((wave      ) * 4 + kf) * FRAG_ELEMS + lane * 16);
                ir  = wmma_f16(ax, b, ir);
                b = *(const v16h*)(fWih + ((8  + wave) * 4 + kf) * FRAG_ELEMS + lane * 16);
                iz  = wmma_f16(ax, b, iz);
                b = *(const v16h*)(fWih + ((16 + wave) * 4 + kf) * FRAG_ELEMS + lane * 16);
                inn = wmma_f16(ax, b, inn);
                b = *(const v16h*)(fWhh + ((wave      ) * 4 + kf) * FRAG_ELEMS + lane * 16);
                hr  = wmma_f16(ah, b, hr);
                b = *(const v16h*)(fWhh + ((8  + wave) * 4 + kf) * FRAG_ELEMS + lane * 16);
                hz  = wmma_f16(ah, b, hz);
                b = *(const v16h*)(fWhh + ((16 + wave) * 4 + kf) * FRAG_ELEMS + lane * 16);
                hn  = wmma_f16(ah, b, hn);
            }
            // PyTorch GRU gate math, elementwise on C layout
            v8f hnew;
#pragma unroll
            for (int g = 0; g < 8; ++g) {
                float r = sigmoidf_(ir[g] + bir + hr[g] + bhr);
                float z = sigmoidf_(iz[g] + biz + hz[g] + bhz);
                float n = tanhf(inn[g] + bin + r * (hn[g] + bhn));
                hnew[g] = (1.f - z) * n + z * h_c[mt][g];
            }
            h_c[mt] = hnew;
        }
        __syncthreads();   // everyone done reading hfrag of step t

        // ---- publish h_{t+1} as A-fragments ----
#pragma unroll
        for (int mt = 0; mt < MT_PER_BLK; ++mt) {
            int base = (mt * 4 + hkf) * FRAG_ELEMS;
#pragma unroll
            for (int g = 0; g < 8; ++g) {
                int m = g + nhalf * 8;
                hfrag[base + afrag_idx(m, hkk)] = (_Float16)h_c[mt][g];
            }
        }
    }
    __syncthreads();

    // ---- write seq_vector (output 1) ----
#pragma unroll
    for (int mt = 0; mt < MT_PER_BLK; ++mt)
#pragma unroll
        for (int g = 0; g < 8; ++g) {
            long long node = node_base + mt * 16 + g + nhalf * 8;
            if (node < N) h_out[node * H_DIM + ucol] = h_c[mt][g];
        }

    // ---- fused GAT linear: x = h_last @ lin_w^T ----
    v8f xa[MT_PER_BLK];
#pragma unroll
    for (int mt = 0; mt < MT_PER_BLK; ++mt) xa[mt] = z8;
#pragma unroll
    for (int kf = 0; kf < 4; ++kf) {
        const v16h bl = *(const v16h*)(flin + (wave * 4 + kf) * FRAG_ELEMS + lane * 16);
#pragma unroll
        for (int mt = 0; mt < MT_PER_BLK; ++mt) {
            const v16h ah = *(const v16h*)(hfrag + (mt * 4 + kf) * FRAG_ELEMS + lane * 16);
            xa[mt] = wmma_f16(ah, bl, xa[mt]);
        }
    }
#pragma unroll
    for (int mt = 0; mt < MT_PER_BLK; ++mt)
#pragma unroll
        for (int g = 0; g < 8; ++g) {
            long long node = node_base + mt * 16 + g + nhalf * 8;
            if (node < N) x_out[node * H_DIM + ucol] = xa[mt][g];
        }
}

// ---- per-node attention logits a_s = x.att_src, a_d = x.att_dst -----------
__global__ void k_dots(const float* __restrict__ x,
                       const float* __restrict__ att_s,
                       const float* __restrict__ att_d,
                       float* __restrict__ as_, float* __restrict__ ad_, int N) {
    int wave = threadIdx.x >> 5, lane = threadIdx.x & 31;
    int node = blockIdx.x * 8 + wave;
    if (node >= N) return;
    float s = 0.f, d = 0.f;
#pragma unroll
    for (int j = 0; j < 4; ++j) {
        float xv = x[(size_t)node * H_DIM + lane + 32 * j];
        s += xv * att_s[lane + 32 * j];
        d += xv * att_d[lane + 32 * j];
    }
#pragma unroll
    for (int off = 16; off; off >>= 1) {
        s += __shfl_xor(s, off, 32);
        d += __shfl_xor(d, off, 32);
    }
    if (lane == 0) { as_[node] = s; ad_[node] = d; }
}

// ---- init: out2 = bias (accumulator base), m = -inf-ish, denom = 0 --------
__global__ void k_init(float* __restrict__ out2, const float* __restrict__ bias,
                       float* __restrict__ m, float* __restrict__ denom, int N) {
    int i = blockIdx.x * 256 + threadIdx.x;
    if (i < N * H_DIM) out2[i] = bias[i & (H_DIM - 1)];
    if (i < N) { m[i] = -3.0e38f; denom[i] = 0.f; }
}

__device__ __forceinline__ void atomicMaxF(float* addr, float v) {
    if (v >= 0.f) atomicMax((int*)addr, __float_as_int(v));
    else          atomicMin((unsigned int*)addr, __float_as_uint(v));
}

// ---- pass 1: alpha + segment max over dst ---------------------------------
__global__ void k_alpha(const int* __restrict__ ei,
                        const float* __restrict__ as_, const float* __restrict__ ad_,
                        float* __restrict__ alpha, float* __restrict__ m,
                        int E, int NE) {
    int e = blockIdx.x * 256 + threadIdx.x;
    if (e >= NE) return;
    int s, d;
    if (e < E) { s = ei[e]; d = ei[E + e]; } else { s = d = e - E; }
    float a = as_[s] + ad_[d];
    a = (a > 0.f) ? a : NEG_SLOPE * a;
    alpha[e] = a;
    atomicMaxF(m + d, a);
}

// ---- pass 2: ex = exp(alpha - m[dst]); denom[dst] += ex -------------------
__global__ void k_exp(const int* __restrict__ ei,
                      const float* __restrict__ alpha, const float* __restrict__ m,
                      float* __restrict__ denom, float* __restrict__ ex,
                      int E, int NE) {
    int e = blockIdx.x * 256 + threadIdx.x;
    if (e >= NE) return;
    int d = (e < E) ? ei[E + e] : (e - E);
    float v = __expf(alpha[e] - m[d]);
    ex[e] = v;
    atomicAdd(denom + d, v);
}

// ---- pass 3: out[dst] += coef * x[src]  (one wave32 per edge) -------------
__global__ void k_scatter(const int* __restrict__ ei,
                          const float* __restrict__ x,
                          const float* __restrict__ ex,
                          const float* __restrict__ denom,
                          float* __restrict__ out2, int E, int NE) {
    int wave = threadIdx.x >> 5, lane = threadIdx.x & 31;
    int e = blockIdx.x * 8 + wave;
    if (e >= NE) return;
    int s, d;
    if (e < E) { s = ei[e]; d = ei[E + e]; } else { s = d = e - E; }
    float coef = ex[e] / denom[d];
    float4 xv = *(const float4*)(x + (size_t)s * H_DIM + lane * 4);
    float* o = out2 + (size_t)d * H_DIM + lane * 4;
    atomicAdd(o + 0, xv.x * coef);
    atomicAdd(o + 1, xv.y * coef);
    atomicAdd(o + 2, xv.z * coef);
    atomicAdd(o + 3, xv.w * coef);
}

// ---------------------------------------------------------------------------
extern "C" void kernel_launch(void* const* d_in, const int* in_sizes, int n_in,
                              void* d_out, int out_size, void* d_ws, size_t ws_size,
                              hipStream_t stream) {
    const float* seq   = (const float*)d_in[0];
    const int*   ei    = (const int*)  d_in[1];
    const float* Wih   = (const float*)d_in[2];
    const float* Whh   = (const float*)d_in[3];
    const float* b_ih  = (const float*)d_in[4];
    const float* b_hh  = (const float*)d_in[5];
    const float* lin_w = (const float*)d_in[6];
    const float* att_s = (const float*)d_in[7];
    const float* att_d = (const float*)d_in[8];
    const float* gbias = (const float*)d_in[9];

    const int N  = in_sizes[0] / (T_STEPS * D_IN);
    const int E  = in_sizes[1] / 2;
    const int NE = E + N;

    // workspace carve-out (~16 MB total)
    char* ws = (char*)d_ws;
    _Float16* fWih = (_Float16*)(ws + 0);       // 96 KB
    _Float16* fWhh = (_Float16*)(ws + 98304);   // 96 KB
    _Float16* flin = (_Float16*)(ws + 196608);  // 32 KB
    size_t o = 229376;
    float* xbuf = (float*)(ws + o); o += (size_t)N * H_DIM * 4;
    float* asb  = (float*)(ws + o); o += (size_t)N * 4;
    float* adb  = (float*)(ws + o); o += (size_t)N * 4;
    float* mb   = (float*)(ws + o); o += (size_t)N * 4;
    float* db   = (float*)(ws + o); o += (size_t)N * 4;
    float* alb  = (float*)(ws + o); o += (size_t)NE * 4;
    float* exb  = (float*)(ws + o); o += (size_t)NE * 4;
    (void)ws_size; (void)n_in; (void)out_size;

    float* out1 = (float*)d_out;                 // seq_vector [N,H]
    float* out2 = out1 + (size_t)N * H_DIM;      // attention_vec [N,H]

    const int prep_elems = 2 * G3 * D_IN + H_DIM * H_DIM;  // 114688
    k_prep<<<(prep_elems + 255) / 256, 256, 0, stream>>>(Wih, Whh, lin_w,
                                                         fWih, fWhh, flin);
    k_gru<<<(N + NODES_PER_BLK - 1) / NODES_PER_BLK, 256, 0, stream>>>(
        seq, fWih, fWhh, flin, b_ih, b_hh, out1, xbuf, N);
    k_dots<<<(N + 7) / 8, 256, 0, stream>>>(xbuf, att_s, att_d, asb, adb, N);
    k_init<<<((size_t)N * H_DIM + 255) / 256, 256, 0, stream>>>(out2, gbias, mb, db, N);
    k_alpha<<<(NE + 255) / 256, 256, 0, stream>>>(ei, asb, adb, alb, mb, E, NE);
    k_exp<<<(NE + 255) / 256, 256, 0, stream>>>(ei, alb, mb, db, exb, E, NE);
    k_scatter<<<(NE + 7) / 8, 256, 0, stream>>>(ei, xbuf, exb, db, out2, E, NE);
}
